// LinearAttention_83004537962800
// MI455X (gfx1250) — matrix operand order repaired and checked by reading, not compiled, not run on hardware
//
#include <hip/hip_runtime.h>
#include <stdint.h>
#include <stddef.h>

// ---------------------------------------------------------------------------
// MI455X (gfx1250) linear-attention block, bf16 WMMA pipeline.
// Stages (all on `stream`, ordered by stream dependencies):
//   0) convert w_qkv / w_out to bf16
//   1) GroupNorm(32 groups) -> xn^T  [b][n][c]  (bf16, B-operand layout)
//   2) QKV GEMM (768x256x4096/batch) + bias + elu+1, scatter to
//        q^T [b][h][n][d], k [b][h][d][n], v [b][h][d][n]   (bf16)
//   3) kv = k . v^T (64x64, K=4096) + k_sum                  (f32)
//   4) out^T[n][e] = q^T . kv^T (K=64), /(denom+eps) -> attn^T [b][n][c]
//   5) y = W_out . attn + b_out + x                          (f32 out)
// ---------------------------------------------------------------------------

typedef __bf16 bf16;
typedef __attribute__((ext_vector_type(16))) __bf16 v16bf;
typedef __attribute__((ext_vector_type(8)))  __bf16 v8bf;
typedef __attribute__((ext_vector_type(8)))  float  v8f;

#define DEV __device__ __forceinline__

static constexpr int Bn = 16;     // batch
static constexpr int C  = 256;    // channels
static constexpr int N  = 4096;   // h*w tokens
static constexpr int H  = 4;      // heads
static constexpr int D  = 64;     // head dim
static constexpr int OC = 768;    // 3*C

DEV bf16 f2bf(float f) {
  uint32_t u = __builtin_bit_cast(uint32_t, f);
  u += 0x7FFFu + ((u >> 16) & 1u);            // round-to-nearest-even
  uint16_t s = (uint16_t)(u >> 16);
  return __builtin_bit_cast(bf16, s);
}
DEV float bf2f(bf16 b) {
  uint16_t s = __builtin_bit_cast(uint16_t, b);
  uint32_t u = ((uint32_t)s) << 16;
  return __builtin_bit_cast(float, u);
}
DEV float elu1(float x) { return x > 0.f ? x + 1.f : __expf(x); }

DEV v8f vzero() { v8f z = {0.f,0.f,0.f,0.f,0.f,0.f,0.f,0.f}; return z; }

DEV v8f wmma_bf16(v16bf a, v16bf b, v8f c) {
  // D = A(16x32) * B(32x16) + C, f32 accumulate
  return __builtin_amdgcn_wmma_f32_16x16x32_bf16(
      /*neg_a=*/false, a, /*neg_b=*/false, b,
      /*c_mod=*/(short)0, c, /*reuse_a=*/false, /*reuse_b=*/false);
}

// A fragment from row-major A[M][lda] at tile (m0,k0).
// ISA layout: lanes 0-15 hold M=lane, K = k0+0..7 and k0+16..23;
//             lanes 16-31 hold K = k0+8..15 and k0+24..31.
DEV v16bf load_a_frag(const bf16* A, int lda, int m0, int k0, int lane) {
  int row = m0 + (lane & 15);
  int kb  = k0 + ((lane & 16) ? 8 : 0);
  const bf16* p = A + (size_t)row * lda + kb;
  v8bf lo = *(const v8bf*)(p);          // K = kb .. kb+7
  v8bf hi = *(const v8bf*)(p + 16);     // K = kb+16 .. kb+23
  v16bf r;
#pragma unroll
  for (int i = 0; i < 8; ++i) { r[i] = lo[i]; r[i + 8] = hi[i]; }
  return r;
}

// B fragment from the TRANSPOSED operand BT[Ncols][ldk] (row n, K contiguous).
// ISA layout: lane L holds column n = n0 + L%16, K = k0 + (L<16 ? 0..7,16..23 : 8..15,24..31).
DEV v16bf load_bt_frag(const bf16* BT, int ldk, int n0, int k0, int lane) {
  int col = n0 + (lane & 15);
  int kb  = k0 + ((lane & 16) ? 8 : 0);
  const bf16* p = BT + (size_t)col * ldk + kb;
  v8bf lo = *(const v8bf*)(p);
  v8bf hi = *(const v8bf*)(p + 16);
  v16bf r;
#pragma unroll
  for (int i = 0; i < 8; ++i) { r[i] = lo[i]; r[i + 8] = hi[i]; }
  return r;
}

// ---------------------------------------------------------------------------
// Stage 0: f32 -> bf16 weight conversion
// ---------------------------------------------------------------------------
__global__ void cvt_weights_kernel(const float* __restrict__ wq,
                                   const float* __restrict__ wo,
                                   bf16* __restrict__ wq_b,
                                   bf16* __restrict__ wo_b) {
  int i = blockIdx.x * blockDim.x + threadIdx.x;
  if (i < OC * C) wq_b[i] = f2bf(wq[i]);
  if (i < C * C)  wo_b[i] = f2bf(wo[i]);
}

// ---------------------------------------------------------------------------
// Stage 1: GroupNorm -> xn^T [b][n][c] bf16. One block per (b, group).
// ---------------------------------------------------------------------------
__global__ __launch_bounds__(256)
void groupnorm_kernel(const float* __restrict__ x,
                      const float* __restrict__ gamma,
                      const float* __restrict__ beta,
                      bf16* __restrict__ xnt) {
  const int GROUPS = 32, CG = C / GROUPS;             // 8 channels per group
  const int b = blockIdx.x / GROUPS, g = blockIdx.x % GROUPS;
  const float* xb = x + ((size_t)b * C + g * CG) * N;
  const int tid = threadIdx.x;

  float s = 0.f, s2 = 0.f;
  for (int i = tid; i < CG * N; i += 256) {
    float v = xb[i];
    s += v; s2 += v * v;
  }
  __shared__ float r0[256], r1[256];
  r0[tid] = s; r1[tid] = s2;
  __syncthreads();
  for (int off = 128; off > 0; off >>= 1) {
    if (tid < off) { r0[tid] += r0[tid + off]; r1[tid] += r1[tid + off]; }
    __syncthreads();
  }
  const float inv = 1.f / (float)(CG * N);
  const float mean = r0[0] * inv;
  const float var  = r1[0] * inv - mean * mean;
  const float rstd = rsqrtf(var + 1e-5f);

  bf16* outb = xnt + (size_t)b * N * C;
  for (int i = tid; i < CG * N; i += 256) {
    int cl = i / N, n = i % N, c = g * CG + cl;
    float v = (xb[i] - mean) * rstd * gamma[c] + beta[c];
    outb[(size_t)n * C + c] = f2bf(v);
  }
}

// ---------------------------------------------------------------------------
// Stage 2: QKV GEMM. Block = 4 waves; wave owns 16(M) x 64(N) strip.
// A = w_qkv[o][c] row-major (bf16), B from xn^T [n][c].
// Fused: +bias, elu+1 on q/k, scatter into consumer-friendly layouts.
// ---------------------------------------------------------------------------
__global__ __launch_bounds__(128)
void qkv_gemm_kernel(const bf16* __restrict__ wq,     // [768][256]
                     const bf16* __restrict__ xnt,    // [B][N][C]
                     const float* __restrict__ bqkv,  // [768]
                     bf16* __restrict__ qt,           // [B][H][N][D]
                     bf16* __restrict__ kbuf,         // [B][H][D][N]
                     bf16* __restrict__ vbuf) {       // [B][H][D][N]
  const int b = blockIdx.z;
  const int m_base = blockIdx.y * 64;                 // over 768
  const int n_base = blockIdx.x * 64;                 // over 4096
  const int wave = threadIdx.x >> 5, lane = threadIdx.x & 31;
  const int m0 = m_base + wave * 16;
  const bf16* BT = xnt + (size_t)b * N * C;

  v8f acc[4];
#pragma unroll
  for (int t = 0; t < 4; ++t) acc[t] = vzero();

  for (int k0 = 0; k0 < C; k0 += 32) {
    if (k0 + 32 < C)
      __builtin_prefetch(BT + (size_t)(n_base + (lane & 15)) * C + k0 + 32, 0, 1);
    v16bf a = load_a_frag(wq, C, m0, k0, lane);
#pragma unroll
    for (int t = 0; t < 4; ++t) {
      v16bf bb = load_bt_frag(BT, C, n_base + 16 * t, k0, lane);
      acc[t] = wmma_bf16(a, bb, acc[t]);
    }
  }

  const int hi8 = (lane & 16) ? 8 : 0;
  const int sec = m_base >> 8;                        // 0=q, 1=k, 2=v
#pragma unroll
  for (int t = 0; t < 4; ++t) {
    int n = n_base + 16 * t + (lane & 15);
#pragma unroll
    for (int r = 0; r < 8; ++r) {
      int o = m0 + r + hi8;
      float val = acc[t][r] + bqkv[o];
      int oc = o & (C - 1), h = oc >> 6, d = oc & (D - 1);
      size_t bh = (size_t)b * H + h;
      if (sec == 0) {
        qt[(bh * N + n) * D + d] = f2bf(elu1(val));
      } else if (sec == 1) {
        kbuf[(bh * D + d) * N + n] = f2bf(elu1(val));
      } else {
        vbuf[(bh * D + d) * N + n] = f2bf(val);
      }
    }
  }
}

// ---------------------------------------------------------------------------
// Stage 3: kv[d][e] = sum_n k[d][n]*v[e][n]  (K=4096), plus k_sum[d].
// One block (4 waves) per (b,h). A = k (row d, n contiguous), BT = v (row e).
// ---------------------------------------------------------------------------
__global__ __launch_bounds__(128)
void kv_ksum_kernel(const bf16* __restrict__ kbuf,
                    const bf16* __restrict__ vbuf,
                    float* __restrict__ kv,           // [B*H][D][D]
                    float* __restrict__ ksum) {       // [B*H][D]
  const int bh = blockIdx.x;
  const bf16* km = kbuf + (size_t)bh * D * N;
  const bf16* vm = vbuf + (size_t)bh * D * N;
  const int wave = threadIdx.x >> 5, lane = threadIdx.x & 31;
  const int m0 = wave * 16;

  v8f acc[4];
#pragma unroll
  for (int t = 0; t < 4; ++t) acc[t] = vzero();

  for (int n0 = 0; n0 < N; n0 += 32) {
    v16bf a = load_a_frag(km, N, m0, n0, lane);
#pragma unroll
    for (int t = 0; t < 4; ++t) {
      v16bf bb = load_bt_frag(vm, N, 16 * t, n0, lane);
      acc[t] = wmma_bf16(a, bb, acc[t]);
    }
  }

  const int hi8 = (lane & 16) ? 8 : 0;
  float* kvp = kv + (size_t)bh * D * D;
#pragma unroll
  for (int t = 0; t < 4; ++t) {
    int e = 16 * t + (lane & 15);
#pragma unroll
    for (int r = 0; r < 8; ++r) {
      int d = m0 + r + hi8;
      kvp[(size_t)d * D + e] = acc[t][r];
    }
  }

  if (threadIdx.x < D) {                              // k_sum rows
    const bf16* kr = km + (size_t)threadIdx.x * N;
    float s = 0.f;
    for (int i = 0; i < N; i += 8) {
      v8bf v = *(const v8bf*)(kr + i);
#pragma unroll
      for (int j = 0; j < 8; ++j) s += bf2f(v[j]);
    }
    ksum[(size_t)bh * D + threadIdx.x] = s;
  }
}

// ---------------------------------------------------------------------------
// Stage 4: out^T[n][e] = sum_d q^T[n][d]*kv[d][e], /(denom[n]+eps).
// One block (8 waves) per (b,h). kv^T(bf16) + denom live in LDS for the
// whole n=4096 loop; B fragments are loaded from LDS once and reused.
// ---------------------------------------------------------------------------
__global__ __launch_bounds__(256)
void apply_kernel(const float* __restrict__ kv,
                  const float* __restrict__ ksum,
                  const bf16* __restrict__ qt,        // [B][H][N][D]
                  bf16* __restrict__ attnT) {         // [B][N][C]
  const int bh = blockIdx.x, b = bh >> 2, h = bh & 3;
  __shared__ bf16  kvT[D][D];                         // kvT[e][d] = kv[d][e]
  __shared__ float ks[D];
  __shared__ float denom[N];
  const int tid = threadIdx.x;

  const float* kvp = kv + (size_t)bh * D * D;
  for (int i = tid; i < D * D; i += 256) {
    int d = i >> 6, e = i & (D - 1);
    kvT[e][d] = f2bf(kvp[i]);
  }
  if (tid < D) ks[tid] = ksum[(size_t)bh * D + tid];
  __syncthreads();

  const bf16* qm = qt + (size_t)bh * N * D;           // [n][d]
  for (int n = tid; n < N; n += 256) {
    const bf16* qr = qm + (size_t)n * D;
    float s = 0.f;
    for (int i = 0; i < D; i += 8) {
      v8bf v = *(const v8bf*)(qr + i);
#pragma unroll
      for (int j = 0; j < 8; ++j) s += bf2f(v[j]) * ks[i + j];
    }
    denom[n] = s + 1e-6f;
  }
  __syncthreads();

  const int wave = tid >> 5, lane = tid & 31;
  v16bf bfr[4][2];
#pragma unroll
  for (int t = 0; t < 4; ++t)
#pragma unroll
    for (int kk = 0; kk < 2; ++kk)
      bfr[t][kk] = load_bt_frag(&kvT[0][0], D, 16 * t, 32 * kk, lane);

  const int hi8 = (lane & 16) ? 8 : 0;
  bf16* outB = attnT + (size_t)b * N * C;
  for (int nt = wave; nt < N / 16; nt += 8) {
    const int n0 = nt * 16;
    v8f acc[4];
#pragma unroll
    for (int t = 0; t < 4; ++t) acc[t] = vzero();
#pragma unroll
    for (int kk = 0; kk < 2; ++kk) {
      v16bf a = load_a_frag(qm, D, n0, 32 * kk, lane);
#pragma unroll
      for (int t = 0; t < 4; ++t) acc[t] = wmma_bf16(a, bfr[t][kk], acc[t]);
    }
#pragma unroll
    for (int t = 0; t < 4; ++t) {
      int c = h * D + 16 * t + (lane & 15);
#pragma unroll
      for (int r = 0; r < 8; ++r) {
        int n = n0 + r + hi8;
        outB[(size_t)n * C + c] = f2bf(acc[t][r] / denom[n]);
      }
    }
  }
}

// ---------------------------------------------------------------------------
// Stage 5: y = W_out * attn + b_out + x. Same shape as stage 2, M=256.
// ---------------------------------------------------------------------------
__global__ __launch_bounds__(128)
void out_proj_kernel(const bf16* __restrict__ wo,     // [256][256]
                     const bf16* __restrict__ attnT,  // [B][N][C]
                     const float* __restrict__ bout,
                     const float* __restrict__ x,
                     float* __restrict__ y) {
  const int b = blockIdx.z;
  const int m_base = blockIdx.y * 64;
  const int n_base = blockIdx.x * 64;
  const int wave = threadIdx.x >> 5, lane = threadIdx.x & 31;
  const int m0 = m_base + wave * 16;
  const bf16* BT = attnT + (size_t)b * N * C;

  v8f acc[4];
#pragma unroll
  for (int t = 0; t < 4; ++t) acc[t] = vzero();

  for (int k0 = 0; k0 < C; k0 += 32) {
    if (k0 + 32 < C)
      __builtin_prefetch(BT + (size_t)(n_base + (lane & 15)) * C + k0 + 32, 0, 1);
    v16bf a = load_a_frag(wo, C, m0, k0, lane);
#pragma unroll
    for (int t = 0; t < 4; ++t) {
      v16bf bb = load_bt_frag(BT, C, n_base + 16 * t, k0, lane);
      acc[t] = wmma_bf16(a, bb, acc[t]);
    }
  }

  const int hi8 = (lane & 16) ? 8 : 0;
#pragma unroll
  for (int t = 0; t < 4; ++t) {
    int n = n_base + 16 * t + (lane & 15);
#pragma unroll
    for (int r = 0; r < 8; ++r) {
      int o = m0 + r + hi8;
      size_t idx = ((size_t)b * C + o) * N + n;
      y[idx] = acc[t][r] + bout[o] + x[idx];          // residual fused
    }
  }
}

// ---------------------------------------------------------------------------
// Launch
// ---------------------------------------------------------------------------
extern "C" void kernel_launch(void* const* d_in, const int* in_sizes, int n_in,
                              void* d_out, int out_size, void* d_ws, size_t ws_size,
                              hipStream_t stream) {
  (void)in_sizes; (void)n_in; (void)out_size; (void)ws_size;
  const float* x     = (const float*)d_in[0];
  const float* gamma = (const float*)d_in[1];
  const float* beta  = (const float*)d_in[2];
  const float* wqkv  = (const float*)d_in[3];
  const float* bqkv  = (const float*)d_in[4];
  const float* wout  = (const float*)d_in[5];
  const float* bout  = (const float*)d_in[6];
  float* y = (float*)d_out;

  char* ws = (char*)d_ws;
  size_t off = 0;
  auto alloc = [&](size_t bytes) -> char* {
    char* p = ws + off;
    off += (bytes + 255) & ~(size_t)255;
    return p;
  };
  bf16*  xnt  = (bf16*)alloc((size_t)Bn * N * C * 2);        // 32 MB
  bf16*  wq_b = (bf16*)alloc((size_t)OC * C * 2);
  bf16*  wo_b = (bf16*)alloc((size_t)C * C * 2);
  bf16*  qt   = (bf16*)alloc((size_t)Bn * H * N * D * 2);    // 32 MB
  bf16*  kb   = (bf16*)alloc((size_t)Bn * H * D * N * 2);    // 32 MB
  bf16*  vb   = (bf16*)alloc((size_t)Bn * H * D * N * 2);    // 32 MB
  float* kv   = (float*)alloc((size_t)Bn * H * D * D * 4);
  float* ks   = (float*)alloc((size_t)Bn * H * D * 4);
  bf16*  attnT = xnt;   // xn^T dead after stage 2; reuse for attn^T

  cvt_weights_kernel<<<(OC * C + 255) / 256, 256, 0, stream>>>(wqkv, wout, wq_b, wo_b);
  groupnorm_kernel<<<Bn * 32, 256, 0, stream>>>(x, gamma, beta, xnt);
  qkv_gemm_kernel<<<dim3(N / 64, OC / 64, Bn), 128, 0, stream>>>(wq_b, xnt, bqkv, qt, kb, vb);
  kv_ksum_kernel<<<Bn * H, 128, 0, stream>>>(kb, vb, kv, ks);
  apply_kernel<<<Bn * H, 256, 0, stream>>>(kv, ks, qt, attnT);
  out_proj_kernel<<<dim3(N / 64, C / 64, Bn), 128, 0, stream>>>(wo_b, attnT, bout, x, y);
}
